// Explicit_corr2_55387898249793
// MI455X (gfx1250) — matrix operand order repaired and checked by reading, not compiled, not run on hardware
//
#include <hip/hip_runtime.h>

// ---------------------------------------------------------------------------
// MI455X (gfx1250) implementation.
//
// Roofline: ~52 GFLOP of small GEMMs vs ~25 MB of unique HBM traffic
// (candidate 13MB + out 9.5MB + weights 0.7MB) -> ~1us of HBM time at
// 23.3 TB/s. Compute-bound => map every layer onto v_wmma_f32_16x16x32_bf16
// (K=32 per op, f32 accumulate). One wave32 = one sample; N=16 spatial
// positions fills exactly one WMMA tile in N.
// ---------------------------------------------------------------------------

typedef __attribute__((ext_vector_type(16))) __bf16 v16bf;
typedef __attribute__((ext_vector_type(8)))  float  v8f;

#define LRELU_SLOPE 0.01f

#define B_BATCH 32
#define C_CH    256
#define KPOS    17          // 17x17 patch positions
#define N_CAND  (B_BATCH * KPOS * KPOS)   // 9248
#define N_SAMP  (B_BATCH + N_CAND)        // 9280

// bf16 weight blob offsets inside d_ws (elements)
#define W0_OFF 0
#define W1_OFF 131072                     // 256*512
#define W2_OFF 163840                     // +128*256
#define W3_OFF 172032                     // +64*128
#define W_TOT  174080                     // +32*64

// LDS strides (elements, multiples of 8 => 16B-aligned b128 LDS ops)
#define S_X0 520   // 512 + pad
#define S_H0 264
#define S_H1 136
#define S_H2 72
#define S_H3 40
#define BUFA_BYTES (16 * S_X0 * 2)        // 16640
#define BUFB_BYTES (16 * S_H0 * 2)        // 8448
#define WAVE_LDS   (BUFA_BYTES + BUFB_BYTES + 64)   // + probs

__device__ __forceinline__ unsigned short f2bf(float f) {
    unsigned int u = __float_as_uint(f);
    u += 0x7FFFu + ((u >> 16) & 1u);      // round-to-nearest-even
    return (unsigned short)(u >> 16);
}
__device__ __forceinline__ float bf2f(unsigned short h) {
    return __uint_as_float(((unsigned int)h) << 16);
}

union ABCast { uint4 q[2]; v16bf v; };

// --------------------------- weight pre-pass -------------------------------
__global__ void convert_weights(const float* __restrict__ w0, const float* __restrict__ w1,
                                const float* __restrict__ w2, const float* __restrict__ w3,
                                unsigned short* __restrict__ ws) {
    int i = blockIdx.x * blockDim.x + threadIdx.x;
    float v;
    if      (i < W1_OFF) v = w0[i];
    else if (i < W2_OFF) v = w1[i - W1_OFF];
    else if (i < W3_OFF) v = w2[i - W2_OFF];
    else if (i < W_TOT)  v = w3[i - W3_OFF];
    else return;
    ws[i] = f2bf(v);
}

// --------------------------- one MLP layer ---------------------------------
// D[OUT,16] = W[OUT,IN] @ X[IN,16] via 16x16x32 bf16 WMMA, LeakyReLU, and
// write-back into the next layer's B-operand layout (position-major X^T).
//
// ISA layouts (CDNA5 ISA):
//  A 16x32 bf16 : lane l (m=l&15, kh=l>>4) holds k = kh*8+{0..7} and
//                 kh*8+16+{0..7}  -> two contiguous 16B runs of row m.
//  B 32x16 bf16 : lane l (n=l&15, kh=l>>4) holds k = kh*8+{0..7} and kh*8+16+{0..7} of
//                 column n        -> two contiguous 16B runs of X^T row n.
//  D 16x16 f32  : acc[r] in lane l = D[r + 8*kh][n].
template<int IN, int OUT, int INS, int OUTS, bool RELU>
__device__ __forceinline__ void mlp_layer(const unsigned short* __restrict__ W,
                                          const unsigned short* src,
                                          unsigned short* dst, int lane) {
    const int n  = lane & 15;
    const int kh = lane >> 4;
    const unsigned short* brow = src + n * INS + 8 * kh;
#pragma unroll 1
    for (int mt = 0; mt < OUT / 16; ++mt) {
        v8f acc = {};
        const unsigned short* wrow = W + (size_t)(mt * 16 + n) * IN + 8 * kh;
#pragma unroll 4
        for (int kk = 0; kk < IN; kk += 32) {
            ABCast a, b;
            a.q[0] = *(const uint4*)(wrow + kk);        // k = kk+8kh .. +7
            a.q[1] = *(const uint4*)(wrow + kk + 16);   // k = kk+8kh+16 .. +23
            b.q[0] = *(const uint4*)(brow + kk);        // k = kk+8kh .. +7
            b.q[1] = *(const uint4*)(brow + kk + 16);   // k = kk+8kh+16 .. +23
            acc = __builtin_amdgcn_wmma_f32_16x16x32_bf16(
                      false, a.v, false, b.v, (short)0, acc, false, false);
        }
        unsigned int pk[4];
#pragma unroll
        for (int i = 0; i < 4; ++i) {
            float lo = acc[2 * i], hi = acc[2 * i + 1];
            if (RELU) {
                lo = lo >= 0.f ? lo : LRELU_SLOPE * lo;
                hi = hi >= 0.f ? hi : LRELU_SLOPE * hi;
            }
            pk[i] = (unsigned)f2bf(lo) | ((unsigned)f2bf(hi) << 16);
        }
        // 8 consecutive m values -> one 16B store into next X^T
        *(uint4*)(dst + n * OUTS + mt * 16 + 8 * kh) = make_uint4(pk[0], pk[1], pk[2], pk[3]);
    }
}

// --------------------------- main kernel -----------------------------------
__global__ void __launch_bounds__(128, 2)
similarity_kernel(const float* __restrict__ tmpl, const float* __restrict__ cand,
                  const unsigned short* __restrict__ wsb, const float* __restrict__ w4,
                  float* __restrict__ out) {
    extern __shared__ char smem[];
    const int lane = threadIdx.x & 31;
    const int wave = threadIdx.x >> 5;
    const int s    = blockIdx.x * 4 + wave;
    if (s >= N_SAMP) return;

    char* wbase = smem + wave * WAVE_LDS;
    unsigned short* bufA = (unsigned short*)wbase;                 // 16 x S_X0
    unsigned short* bufB = (unsigned short*)(wbase + BUFA_BYTES);  // 16 x S_H0
    float* probs = (float*)(wbase + BUFA_BYTES + BUFB_BYTES);      // 16 f32

    const unsigned short* w0b = wsb + W0_OFF;
    const unsigned short* w1b = wsb + W1_OFF;
    const unsigned short* w2b = wsb + W2_OFF;
    const unsigned short* w3b = wsb + W3_OFF;

    // Weight blobs are reused by every wave: pull them toward L2/WGP$ early.
    __builtin_prefetch(w0b, 0, 1);
    __builtin_prefetch(w1b, 0, 1);
    __builtin_prefetch(w2b, 0, 1);
    __builtin_prefetch(w3b, 0, 1);

    const bool isT = (s < B_BATCH);
    int b, y = 0, x = 0;
    if (isT) {
        b = s;
    } else {
        int idx = s - B_BATCH;
        b = idx / (KPOS * KPOS);
        int r = idx % (KPOS * KPOS);
        y = r / KPOS;
        x = r % KPOS;
    }

    // ---- load phase: build X0^T[16][S_X0] bf16 (cols 0..255 mean, 256..511 window)
    for (int c = lane; c < C_CH; c += 32) {
        float v[16];
        if (isT) {
            const float* p = tmpl + (size_t)(b * C_CH + c) * 16;
#pragma unroll
            for (int k = 0; k < 16; ++k) v[k] = p[k];      // aligned -> b128s
        } else {
            const float* p = cand + (size_t)(b * C_CH + c) * 400 + y * 20 + x;
#pragma unroll
            for (int i = 0; i < 4; ++i)
#pragma unroll
                for (int j = 0; j < 4; ++j) v[i * 4 + j] = p[i * 20 + j];
        }
        float m = 0.f;
#pragma unroll
        for (int k = 0; k < 16; ++k) m += v[k];
        m *= (1.f / 16.f);
        const unsigned short mb = f2bf(m);
#pragma unroll
        for (int p = 0; p < 16; ++p) {
            bufA[p * S_X0 + c]         = mb;               // mean rows
            bufA[p * S_X0 + C_CH + c]  = f2bf(v[p]);       // feature rows
        }
    }
    // same-wave LDS producer->consumer: hardware DScnt waits (compiler) order it.

    // ---- MLP: 512 -> 256 -> 128 -> 64 -> 32 (LeakyReLU between) ----
    mlp_layer<512, 256, S_X0, S_H0, true>(w0b, bufA, bufB, lane);
    mlp_layer<256, 128, S_H0, S_H1, true>(w1b, bufB, bufA, lane);
    mlp_layer<128,  64, S_H1, S_H2, true>(w2b, bufA, bufB, lane);
    mlp_layer< 64,  32, S_H2, S_H3, true>(w3b, bufB, bufA, lane);

    // ---- head (32 -> 1) + softmax over the 16 positions ----
    const int p = lane & 15;                 // lanes 16..31 mirror 0..15
    float sc = 0.f;
#pragma unroll
    for (int c = 0; c < 32; ++c) sc += w4[c] * bf2f(bufA[p * S_H3 + c]);

    float mx = sc;
#pragma unroll
    for (int o = 8; o >= 1; o >>= 1) mx = fmaxf(mx, __shfl_xor(mx, o, 16));
    float e = __expf(sc - mx);
    float sum = e;
#pragma unroll
    for (int o = 8; o >= 1; o >>= 1) sum += __shfl_xor(sum, o, 16);
    const float pr = e / sum;
    if (lane < 16) probs[p] = pr;
    asm volatile("s_wait_dscnt 0" ::: "memory");   // make probs visible wave-wide

    float prv[16];
#pragma unroll
    for (int q = 0; q < 16; ++q) prv[q] = probs[q];

    // ---- prob-weighted channel sum (fp32, re-read window from L2) ----
    float* outT = out;                       // [32][272]
    float* outC = out + B_BATCH * 272;       // [32][256][17][17]
    for (int c = lane; c < C_CH; c += 32) {
        float acc = 0.f;
        if (isT) {
            const float* pch = tmpl + (size_t)(b * C_CH + c) * 16;
#pragma unroll
            for (int k = 0; k < 16; ++k) acc += prv[k] * pch[k];
            outT[b * 272 + c] = acc;
        } else {
            const float* pch = cand + (size_t)(b * C_CH + c) * 400 + y * 20 + x;
#pragma unroll
            for (int i = 0; i < 4; ++i)
#pragma unroll
                for (int j = 0; j < 4; ++j) acc += prv[i * 4 + j] * pch[i * 20 + j];
            outC[((size_t)(b * C_CH + c) * KPOS + y) * KPOS + x] = acc;
        }
    }
    if (isT && lane < 16) outT[b * 272 + 256 + p] = pr;   // append wn
}

// --------------------------- launch ----------------------------------------
extern "C" void kernel_launch(void* const* d_in, const int* in_sizes, int n_in,
                              void* d_out, int out_size, void* d_ws, size_t ws_size,
                              hipStream_t stream) {
    const float* tmpl = (const float*)d_in[0];
    const float* cand = (const float*)d_in[1];
    const float* w0   = (const float*)d_in[2];
    const float* w1   = (const float*)d_in[3];
    const float* w2   = (const float*)d_in[4];
    const float* w3   = (const float*)d_in[5];
    const float* w4   = (const float*)d_in[6];
    unsigned short* wsb = (unsigned short*)d_ws;
    float* out = (float*)d_out;

    convert_weights<<<(W_TOT + 255) / 256, 256, 0, stream>>>(w0, w1, w2, w3, wsb);

    const int blocks = (N_SAMP + 3) / 4;                 // 4 waves (samples) / block
    const size_t shmem = 4 * (size_t)WAVE_LDS;           // ~100 KB / WGP workgroup
    similarity_kernel<<<blocks, 128, shmem, stream>>>(tmpl, cand, wsb, w4, out);
}
